// TorchMD_GN_Ext_76020921139240
// MI455X (gfx1250) — matrix operand order repaired and checked by reading, not compiled, not run on hardware
//
#include <hip/hip_runtime.h>
#include <hip/hip_bf16.h>
#include <math.h>

// ---------------- problem constants ----------------
#define NN   10000
#define EE   320000
#define HH   128
#define FF   128
#define RR   50
#define RP   64          // R padded to multiple of 32 for WMMA K
#define LLAYERS 6
#define CUTF 5.0f
#define PIF  3.14159265358979323846f

// ---------------- WMMA types ----------------
typedef _Float16 v16h __attribute__((ext_vector_type(16)));
typedef _Float16 v8h  __attribute__((ext_vector_type(8)));
typedef float    v8f  __attribute__((ext_vector_type(8)));

__device__ __forceinline__ v8f wmma_f16(v16h a, v16h b, v8f c) {
    // D = A(16x32 f16) * B(32x16 f16) + C(16x16 f32)
    return __builtin_amdgcn_wmma_f32_16x16x32_f16(
        /*neg_a=*/false, a, /*neg_b=*/false, b,
        /*c_mod=*/(short)0, c, /*reuse_a=*/false, /*reuse_b=*/false);
}

__device__ __forceinline__ v8f zero8(void) {
    v8f r = {0.f, 0.f, 0.f, 0.f, 0.f, 0.f, 0.f, 0.f};
    return r;
}

__device__ __forceinline__ float silu_f(float x) {
    return x / (1.0f + __expf(-x));
}

// A-fragment: rows m0..m0+15 of a row-major f16 matrix (row stride ldh
// halfwords, ldh % 8 == 0, base 16B aligned).
// lane<16:  elems0-7 = K k0..k0+7,  elems8-15 = K k0+16..k0+23
// lane>=16: elems0-7 = K k0+8..k0+15, elems8-15 = K k0+24..k0+31
__device__ __forceinline__ v16h load_a_frag(const _Float16* base, int ldh,
                                            int m0, int k0, int lane) {
    int row = m0 + (lane & 15);
    int kb  = k0 + ((lane >> 4) << 3);
    const _Float16* p = base + (size_t)row * ldh + kb;
    v8h lo = *(const v8h*)(p);
    v8h hi = *(const v8h*)(p + 16);
    return __builtin_shufflevector(lo, hi, 0,1,2,3,4,5,6,7,8,9,10,11,12,13,14,15);
}

// B-fragment: 32x16 tile of row-major f16 weights (row stride ldh).
// lane L holds column n0+(L&15); K order matches A: lane<16 K k0..k0+7, k0+16..k0+23;
// lane>=16 K k0+8..k0+15, k0+24..k0+31.
__device__ __forceinline__ v16h load_b_frag(const _Float16* base, int ldh,
                                            int k0, int n0, int lane) {
    int kb = k0 + ((lane >> 4) << 3);
    const _Float16* p = base + (size_t)kb * ldh + n0 + (lane & 15);
    v16h r;
    #pragma unroll
    for (int s = 0; s < 8; s++) {
        r[s]     = p[(size_t)s * ldh];
        r[s + 8] = p[(size_t)(s + 16) * ldh];
    }
    return r;
}

// ---------------- kernels ----------------

__global__ void k_zero(float* p, int n) {
    int i = blockIdx.x * blockDim.x + threadIdx.x;
    if (i < n) p[i] = 0.0f;
}

// Per-edge geometry + cosine cutoff + ExpNormal RBF (stored f16, padded to 64)
__global__ void k_edge_rbf(const int* __restrict__ ei, const float* __restrict__ pos,
                           const float* __restrict__ means, const float* __restrict__ betas,
                           float* __restrict__ Cf, float* __restrict__ Cn,
                           _Float16* __restrict__ attr) {
    int e = blockIdx.x * blockDim.x + threadIdx.x;
    if (e >= EE) return;
    int s = ei[e], d = ei[EE + e];
    float dx = pos[3*s+0] - pos[3*d+0];
    float dy = pos[3*s+1] - pos[3*d+1];
    float dz = pos[3*s+2] - pos[3*d+2];
    float dist = sqrtf(dx*dx + dy*dy + dz*dz + 1e-12f);
    float C = (dist < CUTF) ? 0.5f * (__cosf(PIF * dist / CUTF) + 1.0f) : 0.0f;
    Cf[e] = C;
    Cn[e] = (s != d) ? C : 0.0f;
    float ed = __expf(-dist);
    _Float16* row = attr + (size_t)e * RP;
    for (int r = 0; r < RR; r++) {
        float t = ed - means[r];
        row[r] = (_Float16)(C * __expf(-betas[r] * t * t));
    }
    for (int r = RR; r < RP; r++) row[r] = (_Float16)0.0f;
}

// x = emb[z], nbr = nbr_emb[z]
__global__ void k_embed(const int* __restrict__ z, const float* __restrict__ emb,
                        const float* __restrict__ nemb,
                        float* __restrict__ x, float* __restrict__ nbr) {
    int i = blockIdx.x * blockDim.x + threadIdx.x;
    if (i >= NN * HH) return;
    int n = i >> 7, c = i & 127;
    int zz = z[n];
    x[i]   = emb[zz * HH + c];
    nbr[i] = nemb[zz * HH + c];
}

// NeighborEmbedding edge kernel:
//   Wn(64x128) = attr(64x64 f16) @ projW + b, scaled by Cn, then
//   msg[dst] += nbr[src] * Wn   (atomic scatter)
__global__ void __launch_bounds__(128)
k_nbr_edge(const _Float16* __restrict__ attr, const float* __restrict__ Cn,
           const int* __restrict__ ei, const float* __restrict__ W,
           const float* __restrict__ b, const float* __restrict__ nbr,
           float* __restrict__ msg) {
    __shared__ __align__(16) _Float16 Wl[RP * HH];   // 16 KB
    __shared__ float bl[HH];
    __shared__ float WnL[64 * HH];                   // 32 KB
    int tid = threadIdx.x;
    for (int i = tid; i < RP * HH; i += 128) {
        int r = i >> 7;
        Wl[i] = (r < RR) ? (_Float16)W[i] : (_Float16)0.0f;
    }
    if (tid < HH) bl[tid] = b[tid];
    __syncthreads();

    int e0 = blockIdx.x * 64;
    int wave = tid >> 5, lane = tid & 31;
    int m0 = wave * 16;
    const _Float16* Ab = attr + (size_t)e0 * RP;
    v16h a0 = load_a_frag(Ab, RP, m0, 0, lane);
    v16h a1 = load_a_frag(Ab, RP, m0, 32, lane);
    for (int nt = 0; nt < 8; nt++) {
        int n0 = nt * 16;
        v16h b0 = load_b_frag(Wl, HH, 0, n0, lane);
        v16h b1 = load_b_frag(Wl, HH, 32, n0, lane);
        v8f acc = wmma_f16(a0, b0, zero8());
        acc = wmma_f16(a1, b1, acc);
        int col = n0 + (lane & 15);
        int rb  = m0 + ((lane >> 4) << 3);
        float bb = bl[col];
        #pragma unroll
        for (int v = 0; v < 8; v++) {
            int row = rb + v;
            WnL[row * HH + col] = (acc[v] + bb) * Cn[e0 + row];
        }
    }
    __syncthreads();

    int col = tid;  // 0..127
    for (int r = 0; r < 64; r++) {
        int e = e0 + r;
        int s = ei[e], dd = ei[EE + e];
        float v = nbr[(size_t)s * HH + col] * WnL[r * HH + col];
        unsafeAtomicAdd(&msg[(size_t)dd * HH + col], v);
    }
}

// x = concat(x, msg) @ combW(256x128) + b   (in place on x, block-local rows)
__global__ void __launch_bounds__(128)
k_combine(const float* __restrict__ msg, const float* __restrict__ W,
          const float* __restrict__ b, float* __restrict__ x) {
    __shared__ __align__(16) _Float16 Al[64 * 256];  // 32 KB
    __shared__ __align__(16) _Float16 Wl[256 * HH];  // 64 KB
    __shared__ float bl[HH];
    int tid = threadIdx.x;
    for (int i = tid; i < 256 * HH; i += 128) Wl[i] = (_Float16)W[i];
    if (tid < HH) bl[tid] = b[tid];
    int m0g = blockIdx.x * 64;
    for (int i = tid; i < 64 * 256; i += 128) {
        int r = i >> 8, c = i & 255;
        int node = m0g + r;
        float v = 0.0f;
        if (node < NN) v = (c < 128) ? x[(size_t)node * HH + c]
                                     : msg[(size_t)node * HH + (c - 128)];
        Al[i] = (_Float16)v;
    }
    __syncthreads();

    int wave = tid >> 5, lane = tid & 31, m0 = wave * 16;
    v16h a[8];
    #pragma unroll
    for (int kt = 0; kt < 8; kt++) a[kt] = load_a_frag(Al, 256, m0, kt * 32, lane);
    for (int nt = 0; nt < 8; nt++) {
        int n0 = nt * 16;
        v16h bf[8];
        #pragma unroll
        for (int kt = 0; kt < 8; kt++) bf[kt] = load_b_frag(Wl, HH, kt * 32, n0, lane);
        v8f acc = zero8();
        #pragma unroll
        for (int kt = 0; kt < 8; kt++) acc = wmma_f16(a[kt], bf[kt], acc);
        int col = n0 + (lane & 15);
        int rb  = m0 + ((lane >> 4) << 3);
        float bb = bl[col];
        #pragma unroll
        for (int v = 0; v < 8; v++) {
            int node = m0g + rb + v;
            if (node < NN) x[(size_t)node * HH + col] = acc[v] + bb;
        }
    }
}

// Y = X @ W (128x128), no bias
__global__ void __launch_bounds__(128)
k_nodegemm(const float* __restrict__ X, const float* __restrict__ W,
           float* __restrict__ Y) {
    __shared__ __align__(16) _Float16 Al[64 * HH];   // 16 KB
    __shared__ __align__(16) _Float16 Wl[HH * HH];   // 32 KB
    int tid = threadIdx.x;
    for (int i = tid; i < HH * HH; i += 128) Wl[i] = (_Float16)W[i];
    int m0g = blockIdx.x * 64;
    for (int i = tid; i < 64 * HH; i += 128) {
        int r = i >> 7, c = i & 127;
        int node = m0g + r;
        Al[i] = (node < NN) ? (_Float16)X[(size_t)node * HH + c] : (_Float16)0.0f;
    }
    __syncthreads();
    int wave = tid >> 5, lane = tid & 31, m0 = wave * 16;
    v16h a[4];
    #pragma unroll
    for (int kt = 0; kt < 4; kt++) a[kt] = load_a_frag(Al, HH, m0, kt * 32, lane);
    for (int nt = 0; nt < 8; nt++) {
        int n0 = nt * 16;
        v16h bf[4];
        #pragma unroll
        for (int kt = 0; kt < 4; kt++) bf[kt] = load_b_frag(Wl, HH, kt * 32, n0, lane);
        v8f acc = zero8();
        #pragma unroll
        for (int kt = 0; kt < 4; kt++) acc = wmma_f16(a[kt], bf[kt], acc);
        int col = n0 + (lane & 15);
        int rb  = m0 + ((lane >> 4) << 3);
        #pragma unroll
        for (int v = 0; v < 8; v++) {
            int node = m0g + rb + v;
            if (node < NN) Y[(size_t)node * HH + col] = acc[v];
        }
    }
}

// Fused CFConv edge kernel:
//   h1 = silu(attr @ W1 + b1); Wf = (h1 @ W2 + b2) * C;
//   macc[dst] += Y[src] * Wf
__global__ void __launch_bounds__(128)
k_inter_edge(const _Float16* __restrict__ attr, const float* __restrict__ Cf,
             const int* __restrict__ ei,
             const float* __restrict__ W1, const float* __restrict__ b1,
             const float* __restrict__ W2, const float* __restrict__ b2,
             const float* __restrict__ Y, float* __restrict__ macc) {
    __shared__ __align__(16) _Float16 W1l[RP * HH];  // 16 KB
    __shared__ __align__(16) _Float16 W2l[HH * HH];  // 32 KB
    __shared__ __align__(16) _Float16 H1l[64 * HH];  // 16 KB
    __shared__ float b1l[HH], b2l[HH];
    __shared__ float WfL[64 * HH];                   // 32 KB
    int tid = threadIdx.x;
    for (int i = tid; i < RP * HH; i += 128) {
        int r = i >> 7;
        W1l[i] = (r < RR) ? (_Float16)W1[i] : (_Float16)0.0f;
    }
    for (int i = tid; i < HH * HH; i += 128) W2l[i] = (_Float16)W2[i];
    if (tid < HH) { b1l[tid] = b1[tid]; b2l[tid] = b2[tid]; }
    __syncthreads();

    int e0 = blockIdx.x * 64;
    int wave = tid >> 5, lane = tid & 31, m0 = wave * 16;
    const _Float16* Ab = attr + (size_t)e0 * RP;
    v16h a0 = load_a_frag(Ab, RP, m0, 0, lane);
    v16h a1 = load_a_frag(Ab, RP, m0, 32, lane);
    for (int nt = 0; nt < 8; nt++) {
        int n0 = nt * 16;
        v16h b0 = load_b_frag(W1l, HH, 0, n0, lane);
        v16h b1f = load_b_frag(W1l, HH, 32, n0, lane);
        v8f acc = wmma_f16(a0, b0, zero8());
        acc = wmma_f16(a1, b1f, acc);
        int col = n0 + (lane & 15);
        int rb  = m0 + ((lane >> 4) << 3);
        float bb = b1l[col];
        #pragma unroll
        for (int v = 0; v < 8; v++)
            H1l[(rb + v) * HH + col] = (_Float16)silu_f(acc[v] + bb);
    }
    __syncthreads();

    v16h h[4];
    #pragma unroll
    for (int kt = 0; kt < 4; kt++) h[kt] = load_a_frag(H1l, HH, m0, kt * 32, lane);
    for (int nt = 0; nt < 8; nt++) {
        int n0 = nt * 16;
        v16h bf[4];
        #pragma unroll
        for (int kt = 0; kt < 4; kt++) bf[kt] = load_b_frag(W2l, HH, kt * 32, n0, lane);
        v8f acc = zero8();
        #pragma unroll
        for (int kt = 0; kt < 4; kt++) acc = wmma_f16(h[kt], bf[kt], acc);
        int col = n0 + (lane & 15);
        int rb  = m0 + ((lane >> 4) << 3);
        float bb = b2l[col];
        #pragma unroll
        for (int v = 0; v < 8; v++) {
            int row = rb + v;
            WfL[row * HH + col] = (acc[v] + bb) * Cf[e0 + row];
        }
    }
    __syncthreads();

    int col = tid;  // 0..127
    for (int r = 0; r < 64; r++) {
        int e = e0 + r;
        int s = ei[e], dd = ei[EE + e];
        float v = Y[(size_t)s * HH + col] * WfL[r * HH + col];
        unsafeAtomicAdd(&macc[(size_t)dd * HH + col], v);
    }
}

// t = silu(macc @ convW2 + b2); X += t @ linW + lb
__global__ void __launch_bounds__(128)
k_inter_node(const float* __restrict__ Macc,
             const float* __restrict__ W2, const float* __restrict__ b2,
             const float* __restrict__ Wlin, const float* __restrict__ blin,
             float* __restrict__ X) {
    __shared__ __align__(16) _Float16 Al[64 * HH];   // 16 KB
    __shared__ __align__(16) _Float16 W2l[HH * HH];  // 32 KB
    __shared__ __align__(16) _Float16 WLl[HH * HH];  // 32 KB
    __shared__ __align__(16) _Float16 Tl[64 * HH];   // 16 KB
    __shared__ float b2l[HH], bll[HH];
    int tid = threadIdx.x;
    for (int i = tid; i < HH * HH; i += 128) { W2l[i] = (_Float16)W2[i]; WLl[i] = (_Float16)Wlin[i]; }
    if (tid < HH) { b2l[tid] = b2[tid]; bll[tid] = blin[tid]; }
    int m0g = blockIdx.x * 64;
    for (int i = tid; i < 64 * HH; i += 128) {
        int r = i >> 7, c = i & 127;
        int node = m0g + r;
        Al[i] = (node < NN) ? (_Float16)Macc[(size_t)node * HH + c] : (_Float16)0.0f;
    }
    __syncthreads();

    int wave = tid >> 5, lane = tid & 31, m0 = wave * 16;
    v16h a[4];
    #pragma unroll
    for (int kt = 0; kt < 4; kt++) a[kt] = load_a_frag(Al, HH, m0, kt * 32, lane);
    for (int nt = 0; nt < 8; nt++) {
        int n0 = nt * 16;
        v16h bf[4];
        #pragma unroll
        for (int kt = 0; kt < 4; kt++) bf[kt] = load_b_frag(W2l, HH, kt * 32, n0, lane);
        v8f acc = zero8();
        #pragma unroll
        for (int kt = 0; kt < 4; kt++) acc = wmma_f16(a[kt], bf[kt], acc);
        int col = n0 + (lane & 15);
        int rb  = m0 + ((lane >> 4) << 3);
        float bb = b2l[col];
        #pragma unroll
        for (int v = 0; v < 8; v++)
            Tl[(rb + v) * HH + col] = (_Float16)silu_f(acc[v] + bb);
    }
    __syncthreads();

    v16h t[4];
    #pragma unroll
    for (int kt = 0; kt < 4; kt++) t[kt] = load_a_frag(Tl, HH, m0, kt * 32, lane);
    for (int nt = 0; nt < 8; nt++) {
        int n0 = nt * 16;
        v16h bf[4];
        #pragma unroll
        for (int kt = 0; kt < 4; kt++) bf[kt] = load_b_frag(WLl, HH, kt * 32, n0, lane);
        v8f acc = zero8();
        #pragma unroll
        for (int kt = 0; kt < 4; kt++) acc = wmma_f16(t[kt], bf[kt], acc);
        int col = n0 + (lane & 15);
        int rb  = m0 + ((lane >> 4) << 3);
        float bb = bll[col];
        #pragma unroll
        for (int v = 0; v < 8; v++) {
            int node = m0g + rb + v;
            if (node < NN) X[(size_t)node * HH + col] += acc[v] + bb;
        }
    }
}

// ---------------- launch ----------------
extern "C" void kernel_launch(void* const* d_in, const int* in_sizes, int n_in,
                              void* d_out, int out_size, void* d_ws, size_t ws_size,
                              hipStream_t stream) {
    const int*   z     = (const int*)d_in[0];
    const float* pos   = (const float*)d_in[1];
    const int*   ei    = (const int*)d_in[3];
    const float* emb   = (const float*)d_in[4];
    const float* nemb  = (const float*)d_in[5];
    const float* npW   = (const float*)d_in[6];
    const float* npb   = (const float*)d_in[7];
    const float* ncW   = (const float*)d_in[8];
    const float* ncb   = (const float*)d_in[9];
    const float* means = (const float*)d_in[10];
    const float* betas = (const float*)d_in[11];
    const float* mW1   = (const float*)d_in[12];
    const float* mb1   = (const float*)d_in[13];
    const float* mW2   = (const float*)d_in[14];
    const float* mb2   = (const float*)d_in[15];
    const float* cW1   = (const float*)d_in[16];
    const float* cW2   = (const float*)d_in[17];
    const float* cb2   = (const float*)d_in[18];
    const float* lW    = (const float*)d_in[19];
    const float* lb    = (const float*)d_in[20];
    float* X = (float*)d_out;

    char* ws = (char*)d_ws;
    _Float16* attr = (_Float16*)ws; ws += (size_t)EE * RP * sizeof(_Float16);
    float* Cfb = (float*)ws;        ws += (size_t)EE * sizeof(float);
    float* Cnb = (float*)ws;        ws += (size_t)EE * sizeof(float);
    float* nbr = (float*)ws;        ws += (size_t)NN * HH * sizeof(float);
    float* msg = (float*)ws;        ws += (size_t)NN * HH * sizeof(float);  // also macc
    float* Y   = (float*)ws;        ws += (size_t)NN * HH * sizeof(float);

    const int nodeTiles = (NN + 63) / 64;
    const int edgeTiles = EE / 64;

    k_edge_rbf<<<(EE + 255) / 256, 256, 0, stream>>>(ei, pos, means, betas, Cfb, Cnb, attr);
    k_zero<<<(NN * HH + 255) / 256, 256, 0, stream>>>(msg, NN * HH);
    k_embed<<<(NN * HH + 255) / 256, 256, 0, stream>>>(z, emb, nemb, X, nbr);
    k_nbr_edge<<<edgeTiles, 128, 0, stream>>>(attr, Cnb, ei, npW, npb, nbr, msg);
    k_combine<<<nodeTiles, 128, 0, stream>>>(msg, ncW, ncb, X);

    for (int l = 0; l < LLAYERS; l++) {
        k_zero<<<(NN * HH + 255) / 256, 256, 0, stream>>>(msg, NN * HH);
        k_nodegemm<<<nodeTiles, 128, 0, stream>>>(X, cW1 + (size_t)l * HH * FF, Y);
        k_inter_edge<<<edgeTiles, 128, 0, stream>>>(
            attr, Cfb, ei,
            mW1 + (size_t)l * RR * FF, mb1 + (size_t)l * FF,
            mW2 + (size_t)l * FF * FF, mb2 + (size_t)l * FF,
            Y, msg);
        k_inter_node<<<nodeTiles, 128, 0, stream>>>(
            msg, cW2 + (size_t)l * FF * HH, cb2 + (size_t)l * HH,
            lW + (size_t)l * HH * HH, lb + (size_t)l * HH, X);
    }
}
